// KAN_GAT_65592740544645
// MI455X (gfx1250) — matrix operand order repaired and checked by reading, not compiled, not run on hardware
//
#include <hip/hip_runtime.h>

// ---------------------------------------------------------------------------
// KAN-GAT fused pipeline for gfx1250 (MI455X), wave32, WMMA bf16.
// ---------------------------------------------------------------------------

#define DEV __device__ __forceinline__

#if defined(__HIP_DEVICE_COMPILE__) && __has_builtin(__builtin_amdgcn_sched_barrier)
#define SCHED_FENCE() __builtin_amdgcn_sched_barrier(0)
#else
#define SCHED_FENCE()
#endif

static constexpr int N_    = 50000;
static constexpr int E_    = 800000;
static constexpr int IN_   = 128;
static constexpr int HID_  = 64;
static constexpr int EL_   = E_ + N_;       // edges incl. self loops
static constexpr int XC_LD = 200;           // padded stride of concat features
static constexpr int FIN0  = 3 * HID_ + 3;  // 195

typedef __attribute__((ext_vector_type(16))) __bf16 v16bf;
typedef __attribute__((ext_vector_type(8)))  float  v8f;

struct alignas(16) Chunk { unsigned u[4]; };
union BF16x16 { v16bf v; Chunk c[2]; };
union BFRaw   { Chunk c; unsigned short s[8]; };

DEV unsigned short f2bf(float f) {
    unsigned u = __float_as_uint(f);
    u += 0x7FFFu + ((u >> 16) & 1u);   // round-to-nearest-even
    return (unsigned short)(u >> 16);
}
DEV float bf2f(unsigned short s) { return __uint_as_float(((unsigned)s) << 16); }
DEV unsigned fflip(float f) {
    unsigned u = __float_as_uint(f);
    return (u & 0x80000000u) ? ~u : (u | 0x80000000u);
}
DEV float funflip(unsigned e) {
    return (e & 0x80000000u) ? __uint_as_float(e & 0x7FFFFFFFu) : __uint_as_float(~e);
}
DEV float lrelu(float v) { return v > 0.f ? v : 0.2f * v; }

DEV v8f wmma_bf16(v16bf a, v16bf b, v8f c) {
    return __builtin_amdgcn_wmma_f32_16x16x32_bf16(false, a, false, b, (short)0, c,
                                                   false, false);
}

// ------------------------------- weight prep -------------------------------

__global__ void k_bf16cvt(const float* __restrict__ in, unsigned short* __restrict__ out,
                          int n) {
    int i = blockIdx.x * blockDim.x + threadIdx.x;
    if (i < n) out[i] = f2bf(in[i]);
}

// W_g [256][128] row-major  ->  wt [3][256][128] N-major bf16
__global__ void k_wcatT(const float* __restrict__ w0, const float* __restrict__ w1,
                        const float* __restrict__ w2, unsigned short* __restrict__ wt) {
    int i = blockIdx.x * blockDim.x + threadIdx.x;
    if (i >= 3 * IN_ * 256) return;
    int g = i / (IN_ * 256);
    int rem = i - g * IN_ * 256;
    int o = rem >> 7, k = rem & 127;
    const float* w = g == 0 ? w0 : (g == 1 ? w1 : w2);
    wt[i] = f2bf(w[(size_t)o * IN_ + k]);
}

// Merge base_w + spline_w*scaler into one N-major bf16 matrix [outpad][kpad].
// Feature order per input i: [silu, b0..b7] -> 9 consecutive K slots.
__global__ void k_prepw(const float* __restrict__ base, const float* __restrict__ spline,
                        const float* __restrict__ scaler, int outc, int fin, int kpad,
                        int outpad, unsigned short* __restrict__ wt) {
    int i = blockIdx.x * blockDim.x + threadIdx.x;
    if (i >= kpad * outpad) return;
    int o = i / kpad, k = i - o * kpad;
    float v = 0.f;
    if (o < outc && k < fin * 9) {
        int ii = k / 9, j = k - ii * 9;
        v = (j == 0) ? base[o * fin + ii]
                     : spline[(size_t)(o * fin + ii) * 8 + (j - 1)] * scaler[o * fin + ii];
    }
    wt[i] = f2bf(v);
}

// ------------------------- GAT projection (WMMA) ---------------------------
// xp[g][n][h*64+c] = sum_k x[n][k] * W_g[h*64+c][k], bf16 out.
// Scheduling fences keep all 8 B loads in flight under a single wait, then
// the 4 WMMAs issue back-to-back on the accumulation chain.
__global__ void k_proj(const unsigned short* __restrict__ xbf,
                       const unsigned short* __restrict__ wt,   // [3][256][128] N-major
                       unsigned short* __restrict__ xp) {       // [3][N][256]
    int lane = threadIdx.x & 31;
    int wave = threadIdx.x >> 5;
    int r = lane & 15, hh = lane >> 4;
    int row = blockIdx.x * 16 + r;

    const Chunk* xr = (const Chunk*)(xbf + (size_t)row * IN_);
    BF16x16 A[4];
#pragma unroll
    for (int kk = 0; kk < 4; kk++) {               // K = 128 = 4 x 32
        A[kk].c[0] = xr[kk * 4 + hh];              // K chunk k0+8*hh
        A[kk].c[1] = xr[kk * 4 + hh + 2];          // K chunk k0+16+8*hh
    }
    for (int t = wave; t < 48; t += 8) {           // 3 graphs x 16 col tiles
        int g = t >> 4;
        int colBase = (t & 15) << 4;
        const unsigned short* wg = wt + (size_t)g * IN_ * 256;

        BF16x16 B[4];
        SCHED_FENCE();
#pragma unroll
        for (int kk = 0; kk < 4; kk++) {
            const Chunk* bp = (const Chunk*)(wg + (size_t)(colBase + r) * IN_ + kk * 32 + 8 * hh);
            B[kk].c[0] = bp[0];
            B[kk].c[1] = bp[2];
        }
        SCHED_FENCE();                              // all 8 loads issued above
        v8f acc = {0.f, 0.f, 0.f, 0.f, 0.f, 0.f, 0.f, 0.f};
#pragma unroll
        for (int kk = 0; kk < 4; kk++) acc = wmma_bf16(A[kk].v, B[kk].v, acc);
        SCHED_FENCE();

        // D: col = colBase + r ; rows = blk*16 + hh*8 + m
        unsigned short* op =
            xp + (size_t)g * N_ * 256 + (size_t)(blockIdx.x * 16 + hh * 8) * 256 + colBase + r;
#pragma unroll
        for (int m = 0; m < 8; m++) op[(size_t)m * 256] = f2bf(acc[m]);
    }
}

// -------------------------- attention coefficients -------------------------
__global__ void k_attn(const unsigned short* __restrict__ xp, const float* __restrict__ s0,
                       const float* __restrict__ s1, const float* __restrict__ s2,
                       const float* __restrict__ d0, const float* __restrict__ d1,
                       const float* __restrict__ d2, float* __restrict__ as_out,
                       float* __restrict__ ad_out) {
    int idx = blockIdx.x * blockDim.x + threadIdx.x;
    if (idx >= 3 * N_ * 4) return;
    int g = idx / (N_ * 4);
    int rem = idx - g * N_ * 4;
    int n = rem >> 2, h = rem & 3;
    const float* as = g == 0 ? s0 : (g == 1 ? s1 : s2);
    const float* ad = g == 0 ? d0 : (g == 1 ? d1 : d2);
    const Chunk* xr = (const Chunk*)(xp + ((size_t)g * N_ + n) * 256 + h * 64);
    float accs = 0.f, accd = 0.f;
#pragma unroll
    for (int cc = 0; cc < 8; cc++) {
        BFRaw raw;
        raw.c = xr[cc];
#pragma unroll
        for (int j = 0; j < 8; j++) {
            float v = bf2f(raw.s[j]);
            int c = cc * 8 + j;
            accs += v * as[h * 64 + c];
            accd += v * ad[h * 64 + c];
        }
    }
    as_out[idx] = accs;
    ad_out[idx] = accd;
}

// ------------------------------ init scratch -------------------------------
__global__ void k_zero(unsigned* __restrict__ menc, float* __restrict__ ssum,
                       float* __restrict__ outacc) {
    int idx = blockIdx.x * blockDim.x + threadIdx.x;
    if (idx >= 3 * N_ * 64) return;
    outacc[idx] = 0.f;
    if (idx < 3 * N_ * 4) {
        menc[idx] = 0u;          // <= encoding of every finite float
        ssum[idx] = 0.f;
    }
}

// -------------------------- edge softmax: pass max -------------------------
__global__ void k_edge_max(const int* __restrict__ e0, const int* __restrict__ e1,
                           const int* __restrict__ e2, const float* __restrict__ as_,
                           const float* __restrict__ ad_, unsigned* __restrict__ menc) {
    int idx = blockIdx.x * blockDim.x + threadIdx.x;
    if (idx >= 3 * EL_) return;
    int g = idx / EL_, e = idx - g * EL_;
    const int* ei = g == 0 ? e0 : (g == 1 ? e1 : e2);
    int src, dst;
    if (e < E_) { src = ei[e]; dst = ei[E_ + e]; } else { src = dst = e - E_; }
    size_t sb = ((size_t)g * N_ + src) * 4;
    size_t db = ((size_t)g * N_ + dst) * 4;
#pragma unroll
    for (int h = 0; h < 4; h++) {
        float v = lrelu(as_[sb + h] + ad_[db + h]);
        atomicMax(&menc[db + h], fflip(v));
    }
}

// -------------------------- edge softmax: pass sum -------------------------
__global__ void k_edge_sum(const int* __restrict__ e0, const int* __restrict__ e1,
                           const int* __restrict__ e2, const float* __restrict__ as_,
                           const float* __restrict__ ad_, const unsigned* __restrict__ menc,
                           float* __restrict__ ssum) {
    int idx = blockIdx.x * blockDim.x + threadIdx.x;
    if (idx >= 3 * EL_) return;
    int g = idx / EL_, e = idx - g * EL_;
    const int* ei = g == 0 ? e0 : (g == 1 ? e1 : e2);
    int src, dst;
    if (e < E_) { src = ei[e]; dst = ei[E_ + e]; } else { src = dst = e - E_; }
    size_t sb = ((size_t)g * N_ + src) * 4;
    size_t db = ((size_t)g * N_ + dst) * 4;
#pragma unroll
    for (int h = 0; h < 4; h++) {
        float v = lrelu(as_[sb + h] + ad_[db + h]);
        atomicAdd(&ssum[db + h], __expf(v - funflip(menc[db + h])));
    }
}

// ------------------ edge aggregate (head-mean folded in) -------------------
// One wave per edge: lane = h*8 + cb; shfl-xor reduces over the 4 heads so
// only 64 f32 atomics/edge are issued instead of 256.
__global__ void k_edge_aggr(const int* __restrict__ e0, const int* __restrict__ e1,
                            const int* __restrict__ e2, const unsigned short* __restrict__ xp,
                            const float* __restrict__ as_, const float* __restrict__ ad_,
                            const unsigned* __restrict__ menc, const float* __restrict__ ssum,
                            float* __restrict__ outacc) {
    long long warp = ((long long)blockIdx.x * blockDim.x + threadIdx.x) >> 5;
    if (warp >= (long long)3 * EL_) return;
    int lane = threadIdx.x & 31;
    int g = (int)(warp / EL_);
    int e = (int)(warp - (long long)g * EL_);
    const int* ei = g == 0 ? e0 : (g == 1 ? e1 : e2);
    int src, dst;
    if (e < E_) { src = ei[e]; dst = ei[E_ + e]; } else { src = dst = e - E_; }
    int h = lane >> 3, cb = lane & 7;
    size_t sb = ((size_t)g * N_ + src) * 4 + h;
    size_t db = ((size_t)g * N_ + dst) * 4 + h;
    float v = lrelu(as_[sb] + ad_[db]);
    float alpha = __expf(v - funflip(menc[db])) / ssum[db];

    const unsigned short* xr = xp + ((size_t)g * N_ + src) * 256 + h * 64 + cb * 8;
    BFRaw raw;
    raw.c = *(const Chunk*)xr;
    float acc[8];
#pragma unroll
    for (int j = 0; j < 8; j++) acc[j] = bf2f(raw.s[j]) * alpha;
#pragma unroll
    for (int j = 0; j < 8; j++) {
        acc[j] += __shfl_xor(acc[j], 8, 32);    // reduce over head bit 0
        acc[j] += __shfl_xor(acc[j], 16, 32);   // reduce over head bit 1
    }
    if (h == 0) {
        float* op = outacc + ((size_t)g * N_ + dst) * 64 + cb * 8;
#pragma unroll
        for (int j = 0; j < 8; j++) atomicAdd(op + j, acc[j]);
    }
}

// --------------------------- concat -> xc [N,200] --------------------------
__global__ void k_xc(const float* __restrict__ outacc, const float* __restrict__ b0,
                     const float* __restrict__ b1, const float* __restrict__ b2,
                     const float* __restrict__ al0, const float* __restrict__ al1,
                     const float* __restrict__ al2, float* __restrict__ xc) {
    int idx = blockIdx.x * blockDim.x + threadIdx.x;
    if (idx >= N_ * XC_LD) return;
    int n = idx / XC_LD, j = idx - n * XC_LD;
    float v = 0.f;
    if (j < 192) {
        int g = j >> 6, c = j & 63;
        const float* bias = g == 0 ? b0 : (g == 1 ? b1 : b2);
        v = outacc[((size_t)g * N_ + n) * 64 + c] * 0.25f + bias[c];   // head mean
    } else if (j < 195) {
        float a0 = *al0, a1 = *al1, a2 = *al2;
        float mx = fmaxf(a0, fmaxf(a1, a2));
        float x0 = __expf(a0 - mx), x1 = __expf(a1 - mx), x2 = __expf(a2 - mx);
        float s = x0 + x1 + x2;
        v = (j == 192 ? x0 : (j == 193 ? x1 : x2)) / s;
    }
    xc[idx] = v;
}

// --------------------- fused KAN layer: splines + WMMA ---------------------
// LDS holds the expanded feature tile [16][KPAD] bf16: per input i the 9
// features [silu(x), B0(x)..B7(x)] (cubic B-splines on the efficient-KAN
// grid with knots t_j = -2.2 + 0.4*j). One WMMA GEMM over K=KPAD with the
// global B stream double-buffered so loads overlap the WMMA pipe.
template <int FIN, int KPAD, int OUTC, int OUTPAD, int INSTRIDE, bool RELU>
__global__ void k_kan(const float* __restrict__ xin,
                      const unsigned short* __restrict__ wt,   // [OUTPAD][KPAD] N-major
                      float* __restrict__ out) {               // [N][OUTC]
    extern __shared__ unsigned short feat[];                   // [16][KPAD]
    const int tid = threadIdx.x;                               // blockDim = 128
    const int row0 = blockIdx.x * 16;

    for (int it = tid; it < 16 * FIN; it += 128) {
        int rr = it / FIN, i = it - rr * FIN;
        float x = xin[(size_t)(row0 + rr) * INSTRIDE + i];
        float sil = x / (1.f + __expf(-x));
        float b[11];
#pragma unroll
        for (int j = 0; j < 11; j++) {
            float t0 = -2.2f + 0.4f * j;
            b[j] = (x >= t0 && x < t0 + 0.4f) ? 1.f : 0.f;
        }
#pragma unroll
        for (int p = 1; p <= 3; p++) {
            float inv = 1.f / (0.4f * p);
#pragma unroll
            for (int j = 0; j < 11 - p; j++) {
                float tj = -2.2f + 0.4f * j;
                float tjp = tj + 0.4f * p;
                b[j] = (x - tj) * inv * b[j] + (tjp + 0.4f - x) * inv * b[j + 1];
            }
        }
        unsigned short* fp = feat + rr * KPAD + i * 9;
        fp[0] = f2bf(sil);
#pragma unroll
        for (int k = 0; k < 8; k++) fp[1 + k] = f2bf(b[k]);
    }
    if (KPAD > FIN * 9) {
        constexpr int PAD = KPAD - FIN * 9;
        for (int it = tid; it < 16 * PAD; it += 128) {
            int rr = it / PAD, j = it - rr * PAD;
            feat[rr * KPAD + FIN * 9 + j] = 0;
        }
    }
    __syncthreads();

    int lane = tid & 31, wave = tid >> 5;
    int r = lane & 15, hh = lane >> 4;
    constexpr int NT = (OUTC + 15) / 16;
    constexpr int NK = KPAD / 32;
    for (int t = wave; t < NT; t += 4) {
        int colBase = t * 16;
        // prime the B pipeline
        BF16x16 Bcur, Bnxt;
        {
            const unsigned short* bp = wt + (size_t)(colBase + r) * KPAD + 8 * hh;
            Bcur.c[0] = *(const Chunk*)bp;
            Bcur.c[1] = *(const Chunk*)(bp + 16);
        }
        v8f acc = {0.f, 0.f, 0.f, 0.f, 0.f, 0.f, 0.f, 0.f};
#pragma unroll 2
        for (int kk = 0; kk < NK; kk++) {
            if (kk + 1 < NK) {      // issue next B load before consuming current
                const unsigned short* bp =
                    wt + (size_t)(colBase + r) * KPAD + (kk + 1) * 32 + 8 * hh;
                Bnxt.c[0] = *(const Chunk*)bp;
                Bnxt.c[1] = *(const Chunk*)(bp + 16);
            }
            BF16x16 A;
            const unsigned short* ap = feat + r * KPAD + kk * 32 + 8 * hh;
            A.c[0] = *(const Chunk*)ap;
            A.c[1] = *(const Chunk*)(ap + 16);
            SCHED_FENCE();          // Bnxt + A loads stay above the WMMA
            acc = wmma_bf16(A.v, Bcur.v, acc);
            Bcur = Bnxt;
        }
        int col = colBase + r;
        if (col < OUTC) {
#pragma unroll
            for (int m = 0; m < 8; m++) {
                float v = acc[m];
                if (RELU) v = fmaxf(v, 0.f);
                out[(size_t)(row0 + hh * 8 + m) * OUTC + col] = v;
            }
        }
    }
}

// ---------------------------------------------------------------------------

extern "C" void kernel_launch(void* const* d_in, const int* in_sizes, int n_in,
                              void* d_out, int out_size, void* d_ws, size_t ws_size,
                              hipStream_t stream) {
    (void)in_sizes; (void)n_in; (void)out_size; (void)ws_size;

    const float* x    = (const float*)d_in[0];
    const int*   ei0  = (const int*)d_in[1];
    const int*   ei1  = (const int*)d_in[2];
    const int*   ei2  = (const int*)d_in[3];
    const float* al0  = (const float*)d_in[4];
    const float* al1  = (const float*)d_in[5];
    const float* al2  = (const float*)d_in[6];
    const float* W0   = (const float*)d_in[7];
    const float* as0  = (const float*)d_in[8];
    const float* ad0  = (const float*)d_in[9];
    const float* bi0  = (const float*)d_in[10];
    const float* W1   = (const float*)d_in[11];
    const float* as1  = (const float*)d_in[12];
    const float* ad1  = (const float*)d_in[13];
    const float* bi1  = (const float*)d_in[14];
    const float* W2   = (const float*)d_in[15];
    const float* as2  = (const float*)d_in[16];
    const float* ad2  = (const float*)d_in[17];
    const float* bi2  = (const float*)d_in[18];
    const float* fk0b = (const float*)d_in[19];
    const float* fk0s = (const float*)d_in[20];
    const float* fk0c = (const float*)d_in[21];
    const float* fk1b = (const float*)d_in[22];
    const float* fk1s = (const float*)d_in[23];
    const float* fk1c = (const float*)d_in[24];
    const float* k0b  = (const float*)d_in[25];
    const float* k0s  = (const float*)d_in[26];
    const float* k0c  = (const float*)d_in[27];
    const float* k1b  = (const float*)d_in[28];
    const float* k1s  = (const float*)d_in[29];
    const float* k1c  = (const float*)d_in[30];

    char* p = (char*)d_ws;
    auto carve = [&](size_t bytes) -> void* {
        void* r = (void*)p;
        p += (bytes + 255) & ~((size_t)255);
        return r;
    };
    unsigned short* xbf    = (unsigned short*)carve((size_t)N_ * IN_ * 2);
    unsigned short* wcatT  = (unsigned short*)carve((size_t)3 * IN_ * 256 * 2);
    unsigned short* xp     = (unsigned short*)carve((size_t)3 * N_ * 256 * 2);
    float*          as_    = (float*)carve((size_t)3 * N_ * 4 * 4);
    float*          ad_    = (float*)carve((size_t)3 * N_ * 4 * 4);
    unsigned*       menc   = (unsigned*)carve((size_t)3 * N_ * 4 * 4);
    float*          ssum   = (float*)carve((size_t)3 * N_ * 4 * 4);
    float*          outacc = (float*)carve((size_t)3 * N_ * 64 * 4);
    float*          xc     = (float*)carve((size_t)N_ * XC_LD * 4);
    float*          h1     = (float*)carve((size_t)N_ * 64 * 4);
    float*          h2     = (float*)carve((size_t)N_ * 64 * 4);
    float*          h3     = (float*)carve((size_t)N_ * 32 * 4);
    unsigned short* w_fk0  = (unsigned short*)carve((size_t)1760 * 64 * 2);
    unsigned short* w_fk1  = (unsigned short*)carve((size_t)576 * 64 * 2);
    unsigned short* w_k0   = (unsigned short*)carve((size_t)576 * 32 * 2);
    unsigned short* w_k1   = (unsigned short*)carve((size_t)288 * 16 * 2);

    auto blks = [](long long n, int b) { return (unsigned)((n + b - 1) / b); };

    // weight / input prep
    k_bf16cvt<<<blks((long long)N_ * IN_, 256), 256, 0, stream>>>(x, xbf, N_ * IN_);
    k_wcatT<<<blks(3LL * IN_ * 256, 256), 256, 0, stream>>>(W0, W1, W2, wcatT);
    k_prepw<<<blks(1760LL * 64, 256), 256, 0, stream>>>(fk0b, fk0s, fk0c, 64, FIN0, 1760, 64, w_fk0);
    k_prepw<<<blks(576LL * 64, 256), 256, 0, stream>>>(fk1b, fk1s, fk1c, 64, 64, 576, 64, w_fk1);
    k_prepw<<<blks(576LL * 32, 256), 256, 0, stream>>>(k0b, k0s, k0c, 32, 64, 576, 32, w_k0);
    k_prepw<<<blks(288LL * 16, 256), 256, 0, stream>>>(k1b, k1s, k1c, 1, 32, 288, 16, w_k1);

    // GAT projections (WMMA bf16)
    k_proj<<<N_ / 16, 256, 0, stream>>>(xbf, wcatT, xp);

    // attention coefficients
    k_attn<<<blks(3LL * N_ * 4, 256), 256, 0, stream>>>(xp, as0, as1, as2, ad0, ad1, ad2,
                                                        as_, ad_);
    // segment softmax + aggregation
    k_zero<<<blks(3LL * N_ * 64, 256), 256, 0, stream>>>(menc, ssum, outacc);
    k_edge_max<<<blks(3LL * EL_, 256), 256, 0, stream>>>(ei0, ei1, ei2, as_, ad_, menc);
    k_edge_sum<<<blks(3LL * EL_, 256), 256, 0, stream>>>(ei0, ei1, ei2, as_, ad_, menc, ssum);
    k_edge_aggr<<<blks(3LL * EL_ * 32, 256), 256, 0, stream>>>(ei0, ei1, ei2, xp, as_, ad_,
                                                               menc, ssum, outacc);
    // concat
    k_xc<<<blks((long long)N_ * XC_LD, 256), 256, 0, stream>>>(outacc, bi0, bi1, bi2,
                                                               al0, al1, al2, xc);
    // KAN stack (fused spline expansion + WMMA GEMM)
    k_kan<FIN0, 1760, 64, 64, XC_LD, false>
        <<<N_ / 16, 128, 16 * 1760 * 2, stream>>>(xc, w_fk0, h1);
    k_kan<64, 576, 64, 64, 64, true>
        <<<N_ / 16, 128, 16 * 576 * 2, stream>>>(h1, w_fk1, h2);
    k_kan<64, 576, 32, 32, 64, false>
        <<<N_ / 16, 128, 16 * 576 * 2, stream>>>(h2, w_k0, h3);
    k_kan<32, 288, 1, 16, 32, false>
        <<<N_ / 16, 128, 16 * 288 * 2, stream>>>(h3, w_k1, (float*)d_out);
}
